// lstm_l2o_optimizer_50912542327355
// MI455X (gfx1250) — matrix operand order repaired and checked by reading, not compiled, not run on hardware
//
#include <hip/hip_runtime.h>
#include <hip/hip_bf16.h>

typedef __attribute__((ext_vector_type(16))) _Float16 v16h;
typedef __attribute__((ext_vector_type(8)))  float    v8f;

#define S_LEN 20
#define B_N   131072
#define H_DIM 24
#define G_DIM 96
#define GSTR  100     // padded gate-row stride (dwords)
#define NWAVE 2       // waves per block
#define LOG2E 1.44269504088896340736f

// ---------------- fast transcendentals (v_exp_f32 + v_rcp_f32) ----------------
__device__ __forceinline__ float fast_sigmoid(float x) {
  x = fminf(fmaxf(x, -30.f), 30.f);
  float e = __builtin_amdgcn_exp2f(-LOG2E * x);
  return __builtin_amdgcn_rcpf(1.f + e);
}
__device__ __forceinline__ float fast_tanh(float x) {
  x = fminf(fmaxf(x, -15.f), 15.f);
  float e = __builtin_amdgcn_exp2f(-2.f * LOG2E * x);
  return (1.f - e) * __builtin_amdgcn_rcpf(1.f + e);
}
// wave-internal LDS ordering: DS ops are in-order per wave; this blocks compiler
// reordering and drains DScnt before dependent cross-lane reads.
__device__ __forceinline__ void wave_lds_fence() {
  asm volatile("s_wait_dscnt 0" ::: "memory");
}

// ---------------- prep: pack f32 weights into wave32 WMMA B-fragments ----------
// B-matrix (32x16, f16) per-lane layout: lane L holds column N = L%16;
// lanes 0-15 hold K=0..15 (2 per dword), lanes 16-31 hold K=16..31.
// frag dword index = ((mat*6 + tile)*32 + lane)*8 + r
__global__ void lstm_prep_kernel(
    const float* __restrict__ Whh0, const float* __restrict__ Wih1,
    const float* __restrict__ Whh1,
    const float* __restrict__ bih0, const float* __restrict__ bhh0,
    const float* __restrict__ bih1, const float* __restrict__ bhh1,
    const float* __restrict__ wih0, const float* __restrict__ wlin,
    const float* __restrict__ blin,
    unsigned* __restrict__ fragOut, float* __restrict__ miscOut)
{
  int tid = blockIdx.x * blockDim.x + threadIdx.x;
  if (tid < 4608) {
    int r    = tid & 7;
    int lane = (tid >> 3) & 31;
    int t    = (tid >> 8) % 6;
    int mat  = tid / 1536;
    const float* W = (mat == 0) ? Whh0 : (mat == 1) ? Wih1 : Whh1;
    int n  = 16 * t + (lane & 15);
    int k0 = 16 * (lane >> 4) + 2 * r;
    union { _Float16 h[2]; unsigned u; } cvt;
    cvt.h[0] = (k0     < H_DIM) ? (_Float16)W[n * H_DIM + k0]     : (_Float16)0.f;
    cvt.h[1] = (k0 + 1 < H_DIM) ? (_Float16)W[n * H_DIM + k0 + 1] : (_Float16)0.f;
    fragOut[tid] = cvt.u;
  } else if (tid < 4608 + 313) {
    int p = tid - 4608;
    float v;
    if      (p < 96)  v = bih0[p] + bhh0[p];
    else if (p < 192) v = bih1[p - 96] + bhh1[p - 96];
    else if (p < 288) v = wih0[p - 192];
    else if (p < 312) v = wlin[p - 288];
    else              v = blin[0];
    miscOut[p] = v;
  }
}

// ---------------- fragment builders ----------------
// A-matrix 16x32 f16 layout (contiguous k, as in the B pack): lane L row M=L%16;
// lanes 0-15: V0-7 = K0..15 ; lanes 16-31: V0-3 = K16..23, V4-7 = K24..31 (zero pad)
__device__ __forceinline__ v16h build_a(const float* st, int lane) {
  int m = lane & 15;
  const float* row = st + m * H_DIM;
  v16h a;
  if (lane < 16) {
#pragma unroll
    for (int i = 0; i < 8; ++i) a[i] = (_Float16)row[i];
#pragma unroll
    for (int i = 0; i < 8; ++i) a[8 + i] = (_Float16)row[8 + i];
  } else {
#pragma unroll
    for (int i = 0; i < 8; ++i) a[i] = (_Float16)row[16 + i];
#pragma unroll
    for (int i = 0; i < 8; ++i) a[8 + i] = (_Float16)0.f;
  }
  return a;
}

__device__ __forceinline__ v16h load_b(const unsigned* sfrag, int mt, int lane) {
  const unsigned* p = sfrag + (mt * 32 + lane) * 8;
  union { unsigned u[8]; v16h v; } c;
#pragma unroll
  for (int i = 0; i < 8; ++i) c.u[i] = p[i];
  return c.v;
}

// i,f,g,o update; lane L handles row m=L%16, j = 12*(L/16) .. +11
__device__ __forceinline__ void cell_update(float* gates, float* hs, float* cs, int lane) {
  int m  = lane & 15;
  int jb = 12 * (lane >> 4);
  float* g  = gates + m * GSTR + jb;
  float* hp = hs + m * H_DIM + jb;
  float* cp = cs + m * H_DIM + jb;
#pragma unroll
  for (int q = 0; q < 12; ++q) {
    float iv = g[q], fv = g[24 + q], gv = g[48 + q], ov = g[72 + q];
    float cn = fast_sigmoid(fv) * cp[q] + fast_sigmoid(iv) * fast_tanh(gv);
    cp[q] = cn;
    hp[q] = fast_sigmoid(ov) * fast_tanh(cn);
  }
}

// ---------------- main kernel ----------------
__global__ __launch_bounds__(NWAVE * 32, 1) void lstm_main_kernel(
    const float* __restrict__ x, const float* __restrict__ h0in,
    const float* __restrict__ c0in,
    const unsigned* __restrict__ fragg, const float* __restrict__ miscg,
    float* __restrict__ out)
{
  __shared__ unsigned sfrag[4608];
  __shared__ float smisc[320];
  __shared__ float sstate[NWAVE][4][16 * H_DIM];  // h0, c0, h1, c1
  __shared__ float sgates[NWAVE][16 * GSTR];
  __shared__ float sx[NWAVE][16];

  int tid = threadIdx.x;
  for (int i = tid; i < 4608; i += NWAVE * 32) sfrag[i] = fragg[i];
  for (int i = tid; i < 313;  i += NWAVE * 32) smisc[i] = miscg[i];

  int wave = tid >> 5, lane = tid & 31;
  int cbase = (blockIdx.x * NWAVE + wave) * 16;

  float* h0s = sstate[wave][0];
  float* c0s = sstate[wave][1];
  float* h1s = sstate[wave][2];
  float* c1s = sstate[wave][3];
  float* gts = sgates[wave];

  for (int p = lane; p < 16 * H_DIM; p += 32) {
    h0s[p] = h0in[(size_t)cbase * H_DIM + p];
    h1s[p] = h0in[(size_t)B_N * H_DIM + (size_t)cbase * H_DIM + p];
    c0s[p] = c0in[(size_t)cbase * H_DIM + p];
    c1s[p] = c0in[(size_t)B_N * H_DIM + (size_t)cbase * H_DIM + p];
  }
  __syncthreads();

  const float* sb0  = smisc;        // b_ih0+b_hh0 [96]
  const float* sb1  = smisc + 96;   // b_ih1+b_hh1 [96]
  const float* swi0 = smisc + 192;  // W_ih0 [96]
  const float* swl  = smisc + 288;  // W_lin [24]
  float blin = smisc[312];

  int col = lane & 15, half = lane >> 4;

  for (int s = 0; s < S_LEN; ++s) {
    if (lane < 16) sx[wave][lane] = x[(size_t)s * B_N + cbase + lane];
    wave_lds_fence();
    float xr[8];
#pragma unroll
    for (int r = 0; r < 8; ++r) xr[r] = sx[wave][r + 8 * half];

    // ---------- layer 0: gates = b0 + Wih0*x + h0 @ Whh0^T ----------
    v16h a0 = build_a(h0s, lane);
#pragma unroll
    for (int t = 0; t < 6; ++t) {
      int n = 16 * t + col;
      v16h bw = load_b(sfrag, 0 * 6 + t, lane);
      v8f c;
#pragma unroll
      for (int r = 0; r < 8; ++r) c[r] = sb0[n] + swi0[n] * xr[r];
      c = __builtin_amdgcn_wmma_f32_16x16x32_f16(false, a0, false, bw,
                                                 (short)0, c, false, false);
#pragma unroll
      for (int r = 0; r < 8; ++r) gts[(r + 8 * half) * GSTR + n] = c[r];
    }
    wave_lds_fence();
    cell_update(gts, h0s, c0s, lane);
    wave_lds_fence();

    // ---------- layer 1: gates = b1 + h0' @ Wih1^T + h1 @ Whh1^T ----------
    v16h ain = build_a(h0s, lane);   // h0s now holds out0
    v16h ah1 = build_a(h1s, lane);
#pragma unroll
    for (int t = 0; t < 6; ++t) {
      int n = 16 * t + col;
      v16h bwi = load_b(sfrag, 1 * 6 + t, lane);
      v16h bwh = load_b(sfrag, 2 * 6 + t, lane);
      v8f c;
#pragma unroll
      for (int r = 0; r < 8; ++r) c[r] = sb1[n];
      c = __builtin_amdgcn_wmma_f32_16x16x32_f16(false, ain, false, bwi,
                                                 (short)0, c, false, false);
      c = __builtin_amdgcn_wmma_f32_16x16x32_f16(false, ah1, false, bwh,
                                                 (short)0, c, false, false);
#pragma unroll
      for (int r = 0; r < 8; ++r) gts[(r + 8 * half) * GSTR + n] = c[r];
    }
    wave_lds_fence();
    cell_update(gts, h1s, c1s, lane);
    wave_lds_fence();

    // ---------- projection + y = x * (h1 . W_lin + b_lin) ----------
    if (lane < 16) {
      const float* hr = h1s + lane * H_DIM;
      float acc = blin;
#pragma unroll
      for (int j = 0; j < H_DIM; ++j) acc += hr[j] * swl[j];
      out[(size_t)s * B_N + cbase + lane] = sx[wave][lane] * acc;
    }
    if (s + 1 < S_LEN && lane == 0) {
      __builtin_prefetch(&x[(size_t)(s + 1) * B_N + cbase], 0, 0);  // global_prefetch_b8
    }
  }

  // ---------- final states: out = [y | h0T,h1T | c0T,c1T] ----------
  size_t ho = (size_t)S_LEN * B_N;
  size_t bh = (size_t)B_N * H_DIM;
  for (int p = lane; p < 16 * H_DIM; p += 32) {
    out[ho +          (size_t)cbase * H_DIM + p] = h0s[p];
    out[ho + bh     + (size_t)cbase * H_DIM + p] = h1s[p];
    out[ho + 2 * bh + (size_t)cbase * H_DIM + p] = c0s[p];
    out[ho + 3 * bh + (size_t)cbase * H_DIM + p] = c1s[p];
  }
}

extern "C" void kernel_launch(void* const* d_in, const int* in_sizes, int n_in,
                              void* d_out, int out_size, void* d_ws, size_t ws_size,
                              hipStream_t stream) {
  const float* x    = (const float*)d_in[0];
  const float* h0   = (const float*)d_in[1];
  const float* c0   = (const float*)d_in[2];
  const float* Wih0 = (const float*)d_in[3];
  const float* Whh0 = (const float*)d_in[4];
  const float* bih0 = (const float*)d_in[5];
  const float* bhh0 = (const float*)d_in[6];
  const float* Wih1 = (const float*)d_in[7];
  const float* Whh1 = (const float*)d_in[8];
  const float* bih1 = (const float*)d_in[9];
  const float* bhh1 = (const float*)d_in[10];
  const float* Wlin = (const float*)d_in[11];
  const float* blin = (const float*)d_in[12];

  unsigned* frag = (unsigned*)d_ws;
  float* misc = (float*)d_ws + 4608;

  lstm_prep_kernel<<<dim3((4608 + 313 + 255) / 256), dim3(256), 0, stream>>>(
      Whh0, Wih1, Whh1, bih0, bhh0, bih1, bhh1, Wih0, Wlin, blin, frag, misc);

  lstm_main_kernel<<<dim3(B_N / (NWAVE * 16)), dim3(NWAVE * 32), 0, stream>>>(
      x, h0, c0, frag, misc, (float*)d_out);
}
